// GATv2_51548197486661
// MI455X (gfx1250) — matrix operand order repaired and checked by reading, not compiled, not run on hardware
//
#include <hip/hip_runtime.h>

typedef __attribute__((ext_vector_type(16))) _Float16 v16h;
typedef __attribute__((ext_vector_type(8)))  float    v8f;

#define NN   100000
#define EE   1600000
#define E2   (EE + NN)
#define FF   64
#define HH   2
#define CC   32
#define EDD  32
#define GG   256
#define NCC  10
#define NEG_SLOPE 0.2f

// packed weight fragments: [3 layers][2 mats][4 coltiles][2 kblocks][32 lanes][16 halfs]
#define PACK_ELEMS (3 * 2 * 4 * 2 * 32 * 16)

// ---------- monotone float<->uint encoding for atomicMax on floats ----------
__device__ __forceinline__ unsigned encf(float f) {
  unsigned u = __float_as_uint(f);
  return (u & 0x80000000u) ? ~u : (u | 0x80000000u);
}
__device__ __forceinline__ float decf(unsigned u) {
  return __uint_as_float((u & 0x80000000u) ? (u & 0x7FFFFFFFu) : ~u);
}

// ---------- Phase 0: degree + incoming edge-attr sums (for self-loop fill) ----------
__global__ __launch_bounds__(256)
void k_deg_sum(const float* __restrict__ edge_attr, const int* __restrict__ dstIdx,
               float* __restrict__ deg, float* __restrict__ sumAttr) {
  int gid = blockIdx.x * blockDim.x + threadIdx.x;   // E*32 = 51.2M < 2^31
  if (gid >= EE * EDD) return;
  int e = gid >> 5;
  int c = gid & 31;
  int d = dstIdx[e];
  atomicAdd(&sumAttr[d * EDD + c], edge_attr[gid]);
  if (c == 0) atomicAdd(&deg[d], 1.0f);
}

__global__ __launch_bounds__(256)
void k_loop_attr(float* __restrict__ sumAttr, const float* __restrict__ deg) {
  int gid = blockIdx.x * blockDim.x + threadIdx.x;
  if (gid >= NN * EDD) return;
  int n = gid >> 5;
  sumAttr[gid] = sumAttr[gid] / fmaxf(deg[n], 1.0f);
}

// ---------- one-time weight pre-pack into WMMA-B fragment order (f16) ----------
// B 32x16 f16 tile layout: col = lane&15, per-lane element t ->
// K = kBase + (lane>>4)*8 + t for t<8, K = kBase + 16 + (lane>>4)*8 + (t-8) for t>=8.
__global__ __launch_bounds__(256)
void k_pack_w(const float* __restrict__ Wl, const float* __restrict__ Wr,
              _Float16* __restrict__ pack) {
  int gid = blockIdx.x * blockDim.x + threadIdx.x;
  if (gid >= PACK_ELEMS) return;
  int t    = gid & 15;
  int lane = (gid >> 4) & 31;
  int kb   = (gid >> 9) & 1;
  int ct   = (gid >> 10) & 3;
  int mat  = (gid >> 12) & 1;
  int l    = gid >> 13;
  int half = lane >> 4;
  int r    = lane & 15;
  int k    = kb * 32 + (t < 8 ? half * 8 + t : 16 + half * 8 + (t - 8));
  int col  = ct * 16 + r;
  const float* W = (mat ? Wr : Wl) + l * FF * FF;
  pack[gid] = (_Float16)W[k * FF + col];
}

// ---------- vectorized A-fragment load (two runs of 8 consecutive f32) ----------
// A 16x32 f16 tile: lane (half,r) holds row M=r, K = kBase + {half*8+0..7, 16+half*8+0..7}
__device__ __forceinline__ v16h load_a32v(const float* __restrict__ A, int rowBase,
                                          int kBase, int half, int r) {
  const float* p = A + (size_t)(rowBase + r) * FF + kBase + half * 8;
  float4 q0 = ((const float4*)p)[0];
  float4 q1 = ((const float4*)p)[1];
  float4 q2 = ((const float4*)(p + 16))[0];
  float4 q3 = ((const float4*)(p + 16))[1];
  v16h a;
  a[0]  = (_Float16)q0.x; a[1]  = (_Float16)q0.y; a[2]  = (_Float16)q0.z; a[3]  = (_Float16)q0.w;
  a[4]  = (_Float16)q1.x; a[5]  = (_Float16)q1.y; a[6]  = (_Float16)q1.z; a[7]  = (_Float16)q1.w;
  a[8]  = (_Float16)q2.x; a[9]  = (_Float16)q2.y; a[10] = (_Float16)q2.z; a[11] = (_Float16)q2.w;
  a[12] = (_Float16)q3.x; a[13] = (_Float16)q3.y; a[14] = (_Float16)q3.z; a[15] = (_Float16)q3.w;
  return a;
}

// ---------- dual node transform: xl = h@Wl+bl, xr = h@Wr+br (WMMA f32<-f16) ----------
__global__ __launch_bounds__(256)
void k_xlxr(const float* __restrict__ h,
            const v16h* __restrict__ packWl, const v16h* __restrict__ packWr,
            const float* __restrict__ bl, const float* __restrict__ br,
            float* __restrict__ xl, float* __restrict__ xr) {
  int wave = (blockIdx.x * blockDim.x + threadIdx.x) >> 5;
  if (wave >= NN / 16) return;              // wave-uniform: EXEC stays all-ones
  int lane = threadIdx.x & 31;
  int half = lane >> 4;
  int r    = lane & 15;
  int rowBase = wave * 16;

  v16h a0 = load_a32v(h, rowBase, 0,  half, r);
  v16h a1 = load_a32v(h, rowBase, 32, half, r);

#pragma unroll
  for (int ct = 0; ct < 4; ++ct) {
    int col = ct * 16 + r;
    {
      v8f acc = {};
      v16h b0 = packWl[(ct * 2 + 0) * 32 + lane];
      acc = __builtin_amdgcn_wmma_f32_16x16x32_f16(false, a0, false, b0, (short)0, acc, false, false);
      v16h b1 = packWl[(ct * 2 + 1) * 32 + lane];
      acc = __builtin_amdgcn_wmma_f32_16x16x32_f16(false, a1, false, b1, (short)0, acc, false, false);
      float bv = bl[col];
#pragma unroll
      for (int i = 0; i < 8; ++i) {
        int row = rowBase + half * 8 + i;   // D: VGPR i -> M = half*8+i, N = lane&15
        xl[(size_t)row * FF + col] = acc[i] + bv;
      }
    }
    {
      v8f acc = {};
      v16h b0 = packWr[(ct * 2 + 0) * 32 + lane];
      acc = __builtin_amdgcn_wmma_f32_16x16x32_f16(false, a0, false, b0, (short)0, acc, false, false);
      v16h b1 = packWr[(ct * 2 + 1) * 32 + lane];
      acc = __builtin_amdgcn_wmma_f32_16x16x32_f16(false, a1, false, b1, (short)0, acc, false, false);
      float bv = br[col];
#pragma unroll
      for (int i = 0; i < 8; ++i) {
        int row = rowBase + half * 8 + i;
        xr[(size_t)row * FF + col] = acc[i] + bv;
      }
    }
  }
}

// ---------- fused edge-transform + GATv2 logit + segment max ----------
__global__ __launch_bounds__(256)
void k_logits(const float* __restrict__ xl, const float* __restrict__ xr,
              const float* __restrict__ edge_attr, const float* __restrict__ loop_attr,
              const float* __restrict__ We, const float* __restrict__ att,
              const int* __restrict__ srcIdx, const int* __restrict__ dstIdx,
              float* __restrict__ logits, unsigned* __restrict__ mxEnc) {
  __shared__ float sWe[EDD * FF];   // 8 KB
  __shared__ float sAtt[FF];
  for (int i = threadIdx.x; i < EDD * FF; i += blockDim.x) sWe[i] = We[i];
  if (threadIdx.x < FF) sAtt[threadIdx.x] = att[threadIdx.x];
  __syncthreads();

  int e = blockIdx.x * blockDim.x + threadIdx.x;
  if (e >= E2) return;
  int s, d;
  const float* ea;
  if (e < EE) { s = srcIdx[e]; d = dstIdx[e]; ea = edge_attr + (size_t)e * EDD; }
  else        { s = e - EE;    d = s;         ea = loop_attr + (size_t)s * EDD; }

  float ear[EDD];
#pragma unroll
  for (int k = 0; k < EDD; ++k) ear[k] = ea[k];

  const float* xls = xl + (size_t)s * FF;
  const float* xrd = xr + (size_t)d * FF;
  float lg0 = 0.f, lg1 = 0.f;
#pragma unroll 4
  for (int j = 0; j < FF; ++j) {
    float em = 0.f;
#pragma unroll
    for (int k = 0; k < EDD; ++k) em = fmaf(ear[k], sWe[k * FF + j], em);
    float v = xls[j] + xrd[j] + em;
    v = (v > 0.f) ? v : NEG_SLOPE * v;               // leaky_relu
    if (j < CC) lg0 = fmaf(sAtt[j], v, lg0);
    else        lg1 = fmaf(sAtt[j], v, lg1);
  }
  logits[e * 2 + 0] = lg0;
  logits[e * 2 + 1] = lg1;
  atomicMax(&mxEnc[d * 2 + 0], encf(lg0));
  atomicMax(&mxEnc[d * 2 + 1], encf(lg1));
}

// ---------- exp(logit - max) and segment sum ----------
__global__ __launch_bounds__(256)
void k_ew(const float* __restrict__ logits, const int* __restrict__ dstIdx,
          const unsigned* __restrict__ mxEnc, float* __restrict__ ew,
          float* __restrict__ den) {
  int gid = blockIdx.x * blockDim.x + threadIdx.x;
  if (gid >= E2 * 2) return;
  int e  = gid >> 1;
  int hh = gid & 1;
  int d = (e < EE) ? dstIdx[e] : (e - EE);
  float w = expf(logits[gid] - decf(mxEnc[d * 2 + hh]));
  ew[gid] = w;
  atomicAdd(&den[d * 2 + hh], w);
}

// ---------- weighted message scatter: out[dst] += alpha * xl[src] ----------
__global__ __launch_bounds__(256)
void k_scatter(const float* __restrict__ ew, const float* __restrict__ den,
               const float* __restrict__ xl, const int* __restrict__ srcIdx,
               const int* __restrict__ dstIdx, float* __restrict__ out) {
  long long gid = (long long)blockIdx.x * blockDim.x + threadIdx.x;
  if (gid >= (long long)E2 * FF) return;
  int e = (int)(gid >> 6);
  int j = (int)(gid & 63);
  int hh = j >> 5;
  int s, d;
  if (e < EE) { s = srcIdx[e]; d = dstIdx[e]; } else { s = e - EE; d = s; }
  float alpha = ew[e * 2 + hh] / den[d * 2 + hh];
  atomicAdd(&out[(size_t)d * FF + j], alpha * xl[(size_t)s * FF + j]);
}

// ---------- bias + optional relu into the node-feature buffer ----------
__global__ __launch_bounds__(256)
void k_finalize(const float* __restrict__ out, const float* __restrict__ bias,
                float* __restrict__ hbuf, int doRelu) {
  int gid = blockIdx.x * blockDim.x + threadIdx.x;
  if (gid >= NN * FF) return;
  float v = out[gid] + bias[gid & 63];
  if (doRelu) v = fmaxf(v, 0.f);
  hbuf[gid] = v;
}

// ---------- global mean pool ----------
__global__ __launch_bounds__(256)
void k_pool(const float* __restrict__ hbuf, const int* __restrict__ batch,
            float* __restrict__ pooled, float* __restrict__ cnt) {
  int gid = blockIdx.x * blockDim.x + threadIdx.x;
  if (gid >= NN * FF) return;
  int n = gid >> 6;
  int j = gid & 63;
  int g = batch[n];
  atomicAdd(&pooled[g * FF + j], hbuf[gid]);
  if (j == 0) atomicAdd(&cnt[g], 1.0f);
}

// ---------- classifier: [G,64] @ [64,10] + b ----------
__global__ __launch_bounds__(256)
void k_classify(const float* __restrict__ pooled, const float* __restrict__ cnt,
                const float* __restrict__ W, const float* __restrict__ b,
                float* __restrict__ outp) {
  int gid = blockIdx.x * blockDim.x + threadIdx.x;
  if (gid >= GG * NCC) return;
  int g = gid / NCC;
  int k = gid % NCC;
  float inv = 1.0f / fmaxf(cnt[g], 1.0f);
  float acc = b[k];
#pragma unroll
  for (int j = 0; j < FF; ++j)
    acc = fmaf(pooled[g * FF + j] * inv, W[j * NCC + k], acc);
  outp[gid] = acc;
}

extern "C" void kernel_launch(void* const* d_in, const int* in_sizes, int n_in,
                              void* d_out, int out_size, void* d_ws, size_t ws_size,
                              hipStream_t stream) {
  const float* x         = (const float*)d_in[0];
  const float* edge_attr = (const float*)d_in[1];
  const float* Wl        = (const float*)d_in[2];
  const float* bl        = (const float*)d_in[3];
  const float* Wr        = (const float*)d_in[4];
  const float* br        = (const float*)d_in[5];
  const float* We        = (const float*)d_in[6];
  const float* att       = (const float*)d_in[7];
  const float* bias      = (const float*)d_in[8];
  const float* lin_W     = (const float*)d_in[9];
  const float* lin_b     = (const float*)d_in[10];
  const int*   edge_index= (const int*)d_in[11];
  const int*   batch     = (const int*)d_in[12];
  const int*   srcIdx    = edge_index;        // edge_index[0]
  const int*   dstIdx    = edge_index + EE;   // edge_index[1]
  float* outp = (float*)d_out;

  // workspace carving (~145 MB total)
  char* ws = (char*)d_ws;
  size_t off = 0;
  auto carve = [&](size_t bytes) -> char* {
    char* p = ws + off;
    off = (off + bytes + 255) & ~(size_t)255;
    return p;
  };
  float*    deg       = (float*)   carve((size_t)NN * 4);
  float*    loop_attr = (float*)   carve((size_t)NN * EDD * 4);
  float*    xl        = (float*)   carve((size_t)NN * FF * 4);
  float*    xr        = (float*)   carve((size_t)NN * FF * 4);
  float*    logits    = (float*)   carve((size_t)E2 * 2 * 4);
  float*    ewb       = (float*)   carve((size_t)E2 * 2 * 4);
  unsigned* mxEnc     = (unsigned*)carve((size_t)NN * 2 * 4);
  float*    den       = (float*)   carve((size_t)NN * 2 * 4);
  float*    outacc    = (float*)   carve((size_t)NN * FF * 4);
  float*    hbuf      = (float*)   carve((size_t)NN * FF * 4);
  float*    pooled    = (float*)   carve((size_t)GG * FF * 4);
  float*    cnt       = (float*)   carve((size_t)GG * 4);
  _Float16* packW     = (_Float16*)carve((size_t)PACK_ELEMS * 2);

  const int B = 256;

  // Phase 0: weight pre-pack + self-loop attrs (segment mean of incoming attrs)
  k_pack_w<<<(PACK_ELEMS + B - 1) / B, B, 0, stream>>>(Wl, Wr, packW);
  hipMemsetAsync(deg, 0, (size_t)NN * 4, stream);
  hipMemsetAsync(loop_attr, 0, (size_t)NN * EDD * 4, stream);
  k_deg_sum  <<<(EE * EDD + B - 1) / B, B, 0, stream>>>(edge_attr, dstIdx, deg, loop_attr);
  k_loop_attr<<<(NN * EDD + B - 1) / B, B, 0, stream>>>(loop_attr, deg);

  // 3 GATv2 layers
  for (int l = 0; l < 3; ++l) {
    hipMemsetAsync(mxEnc, 0, (size_t)NN * 2 * 4, stream);  // enc(0) < enc(any finite)
    hipMemsetAsync(den,   0, (size_t)NN * 2 * 4, stream);
    hipMemsetAsync(outacc,0, (size_t)NN * FF * 4, stream);

    const float* hin = (l == 0) ? x : hbuf;
    const v16h* packWl = (const v16h*)packW + (size_t)(l * 2 + 0) * 4 * 2 * 32;
    const v16h* packWr = (const v16h*)packW + (size_t)(l * 2 + 1) * 4 * 2 * 32;
    int waves  = NN / 16;                       // 6250 exact tiles
    int blocks = (waves * 32 + B - 1) / B;
    k_xlxr<<<blocks, B, 0, stream>>>(hin, packWl, packWr, bl + l * FF, br + l * FF, xl, xr);

    k_logits<<<(E2 + B - 1) / B, B, 0, stream>>>(xl, xr, edge_attr, loop_attr,
                                                 We + l * EDD * FF, att + l * FF,
                                                 srcIdx, dstIdx, logits, mxEnc);

    k_ew<<<(E2 * 2 + B - 1) / B, B, 0, stream>>>(logits, dstIdx, mxEnc, ewb, den);

    long long scThreads = (long long)E2 * FF;
    k_scatter<<<(unsigned)((scThreads + B - 1) / B), B, 0, stream>>>(ewb, den, xl,
                                                                     srcIdx, dstIdx, outacc);

    k_finalize<<<(NN * FF + B - 1) / B, B, 0, stream>>>(outacc, bias + l * FF, hbuf,
                                                        (l < 2) ? 1 : 0);
  }

  // pooling + classifier
  hipMemsetAsync(pooled, 0, (size_t)GG * FF * 4, stream);
  hipMemsetAsync(cnt,    0, (size_t)GG * 4, stream);
  k_pool    <<<(NN * FF + B - 1) / B, B, 0, stream>>>(hbuf, batch, pooled, cnt);
  k_classify<<<(GG * NCC + B - 1) / B, B, 0, stream>>>(pooled, cnt, lin_W, lin_b, outp);
}
